// LinearDropout1d_9311489098296
// MI455X (gfx1250) — matrix operand order repaired and checked by reading, not compiled, not run on hardware
//
#include <hip/hip_runtime.h>
#include <hip/hip_bf16.h>

// ---------------------------------------------------------------------------
// SNN Linear layer for MI455X (gfx1250, wave32, WMMA)
//   out_st [128,100,2048], out_sc [128,2048], ann_out [128,2048]
// Main GEMM runs as v_wmma_f32_16x16x32_f16 with spikes staged in LDS as f16
// (exact for {0,1}), weights pre-converted to f16 in workspace. IF scan fused.
// ---------------------------------------------------------------------------

typedef _Float16 half_t;
typedef __attribute__((ext_vector_type(16))) _Float16 v16h;
typedef __attribute__((ext_vector_type(8)))  _Float16 v8h;
typedef __attribute__((ext_vector_type(4)))  _Float16 v4h;
typedef __attribute__((ext_vector_type(8)))  float    v8f;
typedef __attribute__((ext_vector_type(4)))  float    v4f;

#define N_BATCH   128
#define T_STEPS   100
#define D_IN      2048
#define D_OUT     2048
#define M_TILES   7          // ceil(100/16) -> 112 padded rows
#define KCHUNK    128
#define A_STRIDE  136        // halves per LDS row: 128 + 8 pad (kills bank conflicts)
#define POT_STRIDE 113       // floats per LDS column: 112 + 1 pad (odd -> conflict-free)
#define THRESH    1.0f

// ---------------------------------------------------------------------------
// Weight f32 -> f16 conversion (runs once per launch; 8 MB into d_ws)
// ---------------------------------------------------------------------------
__global__ __launch_bounds__(256)
void cvt_w_kernel(const float* __restrict__ w, half_t* __restrict__ wf16, int nElem) {
    int i = (blockIdx.x * blockDim.x + threadIdx.x) * 4;
    if (i < nElem) {
        v4f f = *(const v4f*)(w + i);
        v4h h;
        h[0] = (half_t)f[0]; h[1] = (half_t)f[1];
        h[2] = (half_t)f[2]; h[3] = (half_t)f[3];
        *(v4h*)(wf16 + i) = h;
    }
}

// ---------------------------------------------------------------------------
// Fused GEMM (spikes x W^T) + integrate-and-fire scan.
// Block: one batch sample n (blockIdx.y) x 128 output cols (blockIdx.x).
// 8 waves, each owning a 16-col strip; 7 M-tiles (T padded 100->112).
// ---------------------------------------------------------------------------
__global__ __launch_bounds__(256)
void snn_gemm_scan_kernel(const float*  __restrict__ st,
                          const half_t* __restrict__ wf16,
                          const float*  __restrict__ bias,
                          float* __restrict__ out_st,
                          float* __restrict__ out_sc) {
    // 57,856 B shared: phase 1 aliases as f16 A-tile [112][136],
    // phase 2 as f32 pot  [128 cols][113 t]
    __shared__ float smem[128 * POT_STRIDE];
    half_t* a_lds = (half_t*)smem;

    const int tid  = threadIdx.x;
    const int wave = tid >> 5;
    const int lane = tid & 31;
    const int l16  = lane & 15;
    const int hi   = lane >> 4;          // 0: lanes 0-15, 1: lanes 16-31
    const int n    = blockIdx.y;
    const int colbase = blockIdx.x * 128;
    const int ocol = colbase + wave * 16 + l16;

    const float*  a_base = st  + (size_t)n * T_STEPS * D_IN;
    const half_t* brow   = wf16 + (size_t)ocol * D_IN;   // W row = B column (B = W^T)

    v8f acc[M_TILES];
    {
        v8f z = {};
        #pragma unroll
        for (int m = 0; m < M_TILES; ++m) acc[m] = z;
    }

    const int srow = tid >> 5;           // staging: row 0..7 within 8-row slab
    const int scol = (tid & 31) * 4;     // staging: 4 floats per lane, full 128-wide row

    for (int k0 = 0; k0 < D_IN; k0 += KCHUNK) {
        __builtin_prefetch(brow + k0 + KCHUNK, 0, 3);    // global_prefetch next W chunk
        __syncthreads();
        // ---- stage spike tile [112 rows x 128 k] into LDS, f32 -> f16 ----
        #pragma unroll
        for (int it = 0; it < 14; ++it) {
            int r = it * 8 + srow;                        // 0..111 (t index)
            v4h h4 = {};
            if (r < T_STEPS) {
                v4f f = *(const v4f*)(a_base + (size_t)r * D_IN + k0 + scol);
                h4[0] = (half_t)f[0]; h4[1] = (half_t)f[1];
                h4[2] = (half_t)f[2]; h4[3] = (half_t)f[3];
            }
            *(v4h*)(a_lds + r * A_STRIDE + scol) = h4;    // ds_store_b64
        }
        __syncthreads();

        // ---- 4 WMMA K-steps of 32 over this chunk ----
        #pragma unroll
        for (int ks = 0; ks < KCHUNK; ks += 32) {
            // B fragment: lane holds col ocol, 16 contiguous k (hi selects k half)
            v16h bf = *(const v16h*)(brow + k0 + ks + hi * 16);
            #pragma unroll
            for (int m = 0; m < M_TILES; ++m) {
                const half_t* ap = a_lds + (m * 16 + l16) * A_STRIDE + ks + hi * 16;
                v8h alo = *(const v8h*)(ap);          // k:  hi*16 + 0..7
                v8h ahi = *(const v8h*)(ap + 8);      // k:  hi*16 + 8..15
                v16h af = __builtin_shufflevector(alo, ahi,
                            0,1,2,3,4,5,6,7,8,9,10,11,12,13,14,15);
                acc[m] = __builtin_amdgcn_wmma_f32_16x16x32_f16(
                            false, af, false, bf, (short)0, acc[m], false, false);
            }
        }
    }

    __syncthreads();
    // ---- scatter pot to LDS column-major: pot[col][t] ----
    {
        const int clocal = wave * 16 + l16;
        #pragma unroll
        for (int m = 0; m < M_TILES; ++m) {
            #pragma unroll
            for (int v = 0; v < 8; ++v) {
                // C/D layout: VGPR v holds row M = v + 8*hi within the 16x16 tile
                smem[clocal * POT_STRIDE + m * 16 + v + hi * 8] = acc[m][v];
            }
        }
    }
    __syncthreads();

    // ---- integrate-and-fire scan: one thread per output column ----
    if (tid < 128) {
        const int oc = colbase + tid;
        float p   = bias[oc];
        float cnt = 0.0f;
        float* outp = out_st + (size_t)n * T_STEPS * D_OUT + oc;
        const float* pcol = smem + tid * POT_STRIDE;
        for (int t = 0; t < T_STEPS; ++t) {
            p += pcol[t];
            float s = (p >= THRESH) ? 1.0f : 0.0f;
            p -= s * THRESH;
            outp[(size_t)t * D_OUT] = s;                 // coalesced across 128 threads
            cnt += s;
        }
        out_sc[(size_t)n * D_OUT + oc] = cnt;
    }
}

// ---------------------------------------------------------------------------
// ANN path: relu(sc @ W^T + bias). Tiny GEMM [128 x 2048 x 2048]; bias folded
// into the C initializer, ReLU at store. Same WMMA fragment scheme.
// ---------------------------------------------------------------------------
__global__ __launch_bounds__(256)
void ann_gemm_kernel(const float*  __restrict__ sc,
                     const half_t* __restrict__ wf16,
                     const float*  __restrict__ bias,
                     float* __restrict__ annout) {
    const int tid  = threadIdx.x;
    const int wave = tid >> 5;
    const int lane = tid & 31;
    const int l16  = lane & 15;
    const int hi   = lane >> 4;
    const int ocol = blockIdx.x * 128 + wave * 16 + l16;
    const float bv = bias[ocol];
    const half_t* brow = wf16 + (size_t)ocol * D_IN;

    v8f acc[8];
    #pragma unroll
    for (int m = 0; m < 8; ++m) {
        #pragma unroll
        for (int v = 0; v < 8; ++v) acc[m][v] = bv;   // C = bias broadcast over rows
    }

    for (int k0 = 0; k0 < D_IN; k0 += 32) {
        v16h bf = *(const v16h*)(brow + k0 + hi * 16);
        #pragma unroll
        for (int m = 0; m < 8; ++m) {
            const float* ap = sc + (size_t)(m * 16 + l16) * D_IN + k0 + hi * 16;
            v4f f0 = *(const v4f*)(ap);
            v4f f1 = *(const v4f*)(ap + 4);
            v4f f2 = *(const v4f*)(ap + 8);
            v4f f3 = *(const v4f*)(ap + 12);
            v16h af;
            #pragma unroll
            for (int j = 0; j < 4; ++j) {
                af[j]      = (half_t)f0[j];
                af[j + 4]  = (half_t)f1[j];
                af[j + 8]  = (half_t)f2[j];
                af[j + 12] = (half_t)f3[j];
            }
            acc[m] = __builtin_amdgcn_wmma_f32_16x16x32_f16(
                        false, af, false, bf, (short)0, acc[m], false, false);
        }
    }

    #pragma unroll
    for (int m = 0; m < 8; ++m) {
        #pragma unroll
        for (int v = 0; v < 8; ++v) {
            int rn = m * 16 + v + hi * 8;            // batch row
            float x = acc[m][v];
            annout[(size_t)rn * D_OUT + ocol] = x > 0.0f ? x : 0.0f;
        }
    }
}

// ---------------------------------------------------------------------------
extern "C" void kernel_launch(void* const* d_in, const int* in_sizes, int n_in,
                              void* d_out, int out_size, void* d_ws, size_t ws_size,
                              hipStream_t stream) {
    const float* st   = (const float*)d_in[0];   // [128,100,2048]
    const float* sc   = (const float*)d_in[1];   // [128,2048]
    const float* w    = (const float*)d_in[2];   // [2048,2048]
    const float* bias = (const float*)d_in[3];   // [2048]

    half_t* wf16 = (half_t*)d_ws;                // 8 MB f16 weight copy

    float* out_st = (float*)d_out;                                   // [128,100,2048]
    float* out_sc = out_st + (size_t)N_BATCH * T_STEPS * D_OUT;      // [128,2048]
    float* ann    = out_sc + (size_t)N_BATCH * D_OUT;                // [128,2048]

    // 1) weight conversion: 4M elems / 4 per thread / 256 per block
    cvt_w_kernel<<<(D_OUT * D_IN) / (4 * 256), 256, 0, stream>>>(w, wf16, D_OUT * D_IN);

    // 2) fused spike GEMM + IF scan: grid (colBlocks=16, n=128)
    snn_gemm_scan_kernel<<<dim3(D_OUT / 128, N_BATCH), 256, 0, stream>>>(
        st, wf16, bias, out_st, out_sc);

    // 3) ANN path
    ann_gemm_kernel<<<D_OUT / 128, 256, 0, stream>>>(sc, wf16, bias, ann);
}
